// LocationAwareSpatialAttention_30081950941202
// MI455X (gfx1250) — compile-verified, matrix operand order checked by reading, not run on hardware
//
#include <hip/hip_runtime.h>

#define BB 16
#define NN 1024
#define HH 256
#define HEADS 4
#define HD 64
#define BD 8
#define SCALE_F 0.125f
#define REG_W 1e-5f
#define LN_EPS 1e-5f

typedef __attribute__((ext_vector_type(16))) _Float16 v16h;
typedef __attribute__((ext_vector_type(8)))  _Float16 v8h;
typedef __attribute__((ext_vector_type(8)))  float    v8f;
typedef __attribute__((ext_vector_type(4)))  float    v4f;

union F8 { v8f v; v4f q[2]; };
union H16 { v16h v; v8h h[2]; };

// Load a 16-element f16 WMMA operand fragment for this lane.
// CDNA5 16-bit A/B layout: elements e=0..7 are 8 contiguous halves at p,
// e=8..15 at p+16 (caller bakes the (lane>>4)*8 half-wave K offset into p).
static __device__ inline v16h load_frag(const _Float16* p) {
  H16 u;
  u.h[0] = *(const v8h*)(p);
  u.h[1] = *(const v8h*)(p + 16);
  return u.v;
}

// ---------------------------------------------------------------------------
// Kernel 0: adj_bias[h][n][m] = sum_r u[h][n][r] * v[h][r][m]   (r < 8)
// 16 MB, L2-resident; read vectorized (v4f) by the attention kernel.
// ---------------------------------------------------------------------------
__global__ __launch_bounds__(256) void adj_kernel(const float* __restrict__ u,
                                                  const float* __restrict__ v,
                                                  float* __restrict__ adj) {
  int h = blockIdx.x >> 10;
  int n = blockIdx.x & (NN - 1);
  float ur[BD];
#pragma unroll
  for (int r = 0; r < BD; ++r) ur[r] = u[((size_t)h * NN + n) * BD + r];
#pragma unroll
  for (int mm = 0; mm < 4; ++mm) {
    int m = threadIdx.x + mm * 256;
    float s = 0.f;
#pragma unroll
    for (int r = 0; r < BD; ++r) s += ur[r] * v[((size_t)h * BD + r) * NN + m];
    adj[((size_t)h * NN + n) * NN + m] = s;
  }
}

// ---------------------------------------------------------------------------
// Kernel 1: h = (x+loc+ds) @ W_loc^T + b ; low = h @ W_ql^T + b ;
//           qkv = low @ W_qh^T + b  -> q (f16, pre-scaled), k (f16), v^T (f16)
// 4 tokens per 256-thread block: each W_loc row element is reused 4x.
// ---------------------------------------------------------------------------
__global__ __launch_bounds__(256) void qkv_kernel(
    const float* __restrict__ x, const float* __restrict__ loc_emb,
    const float* __restrict__ ds_emb, const float* __restrict__ W_loc,
    const float* __restrict__ b_loc, const float* __restrict__ W_ql,
    const float* __restrict__ b_ql, const float* __restrict__ W_qh,
    const float* __restrict__ b_qh, _Float16* __restrict__ qb,
    _Float16* __restrict__ kb, _Float16* __restrict__ vtb) {
  __shared__ float hin[4][HH];
  __shared__ float hmid[4][HH];
  __shared__ float lowv[4][3 * BD];
  int t = threadIdx.x;
  int b = blockIdx.x >> 8;          // 256 token-groups per batch
  int n0 = (blockIdx.x & 255) * 4;

#pragma unroll
  for (int s = 0; s < 4; ++s)
    hin[s][t] = x[((size_t)b * NN + n0 + s) * HH + t] +
                loc_emb[(size_t)(n0 + s) * HH + t] + ds_emb[t];
  __syncthreads();

  float a0 = b_loc[t], a1 = a0, a2 = a0, a3 = a0;
  const float* wr = W_loc + (size_t)t * HH;
  for (int j = 0; j < HH; ++j) {
    float w = wr[j];
    a0 += hin[0][j] * w; a1 += hin[1][j] * w;
    a2 += hin[2][j] * w; a3 += hin[3][j] * w;
  }
  hmid[0][t] = a0; hmid[1][t] = a1; hmid[2][t] = a2; hmid[3][t] = a3;
  __syncthreads();

  if (t < 4 * 3 * BD) {             // 96 threads: (token s, row r)
    int s = t / (3 * BD), r = t % (3 * BD);
    float acc = b_ql[r];
    const float* w = W_ql + (size_t)r * HH;
    for (int j = 0; j < HH; ++j) acc += hmid[s][j] * w[j];
    lowv[s][r] = acc;
  }
  __syncthreads();

  int head = t >> 6, hd = t & 63;
#pragma unroll
  for (int s = 0; s < 4; ++s) {
    int n = n0 + s;
    float q = b_qh[t], k = b_qh[HH + t], vv = b_qh[2 * HH + t];
#pragma unroll
    for (int r = 0; r < 3 * BD; ++r) {
      float l = lowv[s][r];
      q  += l * W_qh[(size_t)t * (3 * BD) + r];
      k  += l * W_qh[(size_t)(HH + t) * (3 * BD) + r];
      vv += l * W_qh[(size_t)(2 * HH + t) * (3 * BD) + r];
    }
    size_t hb = ((size_t)(b * HEADS + head) * NN + n) * HD + hd;
    qb[hb] = (_Float16)(q * SCALE_F);   // fold softmax scale into Q (2^-3)
    kb[hb] = (_Float16)k;
    vtb[((size_t)(b * HEADS + head) * HD + hd) * NN + n] = (_Float16)vv;
  }
}

// ---------------------------------------------------------------------------
// Kernel 2: flash attention, transposed-score formulation. 4 waves/block,
// each wave owns 16 query rows; no LDS, no barriers.
//   S^T = K Q^T  -> C-frag has query-row m in the LANE index and key index n
//   in the REGISTER index, so (a) row softmax is a register reduction plus a
//   single shfl_xor(16), and (b) the exp'd S^T C-frags ARE the A-fragment
//   layout for P@V (tile0 regs -> elems 0-7, tile1 regs -> elems 8-15).
// 64-column chunks: 8 wmma for S^T, 8 wmma for P@V.
// ---------------------------------------------------------------------------
__global__ __launch_bounds__(128) void attn_kernel(
    const _Float16* __restrict__ qbuf, const _Float16* __restrict__ kbuf,
    const _Float16* __restrict__ vtbuf, const float* __restrict__ adj,
    float* __restrict__ ctx) {
  int tid = threadIdx.x;
  int wave = tid >> 5, lane = tid & 31;
  int hf = lane >> 4;        // half-wave (K-offset group)
  int l16 = lane & 15;
  int bid = blockIdx.x;
  int mblk = bid & 15;
  int head = (bid >> 4) & 3;
  int b = bid >> 6;

  const _Float16* qh = qbuf + (size_t)(b * HEADS + head) * NN * HD;
  const _Float16* kh = kbuf + (size_t)(b * HEADS + head) * NN * HD;
  const _Float16* vth = vtbuf + (size_t)(b * HEADS + head) * HD * NN;

  int mbase = mblk * 64 + wave * 16;
  int mrow = mbase + l16;                       // this lane's query row
  const float* adjrow = adj + (size_t)head * NN * NN + (size_t)mrow * NN;

  // Q as the B operand of S^T = K Q^T (head_dim halves 0..31 / 32..63)
  v16h bq0 = load_frag(qh + (size_t)mrow * HD + hf * 8);
  v16h bq1 = load_frag(qh + (size_t)mrow * HD + hf * 8 + 32);

  v8f acc0 = {}, acc1 = {}, acc2 = {}, acc3 = {};
  float mold = -3.0e38f;   // running row max   (lane-form: row = l16)
  float lsum = 0.f;        // running row denom (lane-form)

  for (int c = 0; c < NN / 64; ++c) {
    v8f s0, s1, s2, s3;
    // ---- S^T tiles: C input = adj bias (free add), A = K tile, B = Q^T ----
#pragma unroll
    for (int t = 0; t < 4; ++t) {
      int nb = c * 64 + t * 16;
      F8 sb;
      sb.q[0] = *(const v4f*)(adjrow + nb + hf * 8);
      sb.q[1] = *(const v4f*)(adjrow + nb + hf * 8 + 4);
      v16h ak0 = load_frag(kh + (size_t)(nb + l16) * HD + hf * 8);
      v16h ak1 = load_frag(kh + (size_t)(nb + l16) * HD + hf * 8 + 32);
      v8f sv = sb.v;
      sv = __builtin_amdgcn_wmma_f32_16x16x32_f16(false, ak0, false, bq0, (short)0, sv, false, false);
      sv = __builtin_amdgcn_wmma_f32_16x16x32_f16(false, ak1, false, bq1, (short)0, sv, false, false);
      if (t == 0) s0 = sv; else if (t == 1) s1 = sv; else if (t == 2) s2 = sv; else s3 = sv;
    }

    // ---- online softmax over this 64-col chunk (register reduction) ----
    float tmax = s0[0];
#pragma unroll
    for (int i = 0; i < 8; ++i) {
      tmax = fmaxf(tmax, fmaxf(fmaxf(s0[i], s1[i]), fmaxf(s2[i], s3[i])));
    }
    tmax = fmaxf(tmax, __shfl_xor(tmax, 16, 32));   // merge lane halves
    float mnew = fmaxf(mold, tmax);
    float corrl = __expf(mold - mnew);
    mold = mnew;

    float rs = 0.f;
#pragma unroll
    for (int i = 0; i < 8; ++i) {
      float p0 = __expf(s0[i] - mnew); s0[i] = p0;
      float p1 = __expf(s1[i] - mnew); s1[i] = p1;
      float p2 = __expf(s2[i] - mnew); s2[i] = p2;
      float p3 = __expf(s3[i] - mnew); s3[i] = p3;
      rs += (p0 + p1) + (p2 + p3);
    }
    rs += __shfl_xor(rs, 16, 32);
    lsum = lsum * corrl + rs;

    // corr is per query-row (lane-form); accumulators index rows by register
#pragma unroll
    for (int i = 0; i < 8; ++i) {
      float cr = __shfl(corrl, i + 8 * hf, 32);
      acc0[i] *= cr; acc1[i] *= cr; acc2[i] *= cr; acc3[i] *= cr;
    }

    // ---- P fragments directly from S^T C-frags (no LDS) ----
    H16 pa0, pa1;
#pragma unroll
    for (int e = 0; e < 8; ++e) {
      pa0.v[e]     = (_Float16)s0[e];
      pa0.v[e + 8] = (_Float16)s1[e];
      pa1.v[e]     = (_Float16)s2[e];
      pa1.v[e + 8] = (_Float16)s3[e];
    }

    // ---- ctx += P @ V ; V^T layout gives contiguous 16B B-frag loads ----
#pragma unroll
    for (int g = 0; g < 4; ++g) {
      v16h bv0 = load_frag(vth + (size_t)(g * 16 + l16) * NN + c * 64 + hf * 8);
      v16h bv1 = load_frag(vth + (size_t)(g * 16 + l16) * NN + c * 64 + hf * 8 + 32);
      v8f a = (g == 0) ? acc0 : (g == 1) ? acc1 : (g == 2) ? acc2 : acc3;
      a = __builtin_amdgcn_wmma_f32_16x16x32_f16(false, pa0.v, false, bv0, (short)0, a, false, false);
      a = __builtin_amdgcn_wmma_f32_16x16x32_f16(false, pa1.v, false, bv1, (short)0, a, false, false);
      if (g == 0) acc0 = a; else if (g == 1) acc1 = a; else if (g == 2) acc2 = a; else acc3 = a;
    }
  }

  // normalize (denominator lane->register transpose) and write ctx slice
#pragma unroll
  for (int i = 0; i < 8; ++i) {
    float linv = 1.0f / __shfl(lsum, i + 8 * hf, 32);
    int mg = mbase + i + 8 * hf;
    float* crow = ctx + ((size_t)b * NN + mg) * HH + head * HD;
    crow[0 + l16]  = acc0[i] * linv;
    crow[16 + l16] = acc1[i] * linv;
    crow[32 + l16] = acc2[i] * linv;
    crow[48 + l16] = acc3[i] * linv;
  }
}

// ---------------------------------------------------------------------------
// Kernel 3: out = LN(ctx @ W_ol^T @ W_oh^T + b + residual); deterministic
// LDS tree reductions. One 256-thread block per token.
// ---------------------------------------------------------------------------
__global__ __launch_bounds__(256) void outln_kernel(
    const float* __restrict__ ctx, const float* __restrict__ x,
    const float* __restrict__ W_ol, const float* __restrict__ b_ol,
    const float* __restrict__ W_oh, const float* __restrict__ b_oh,
    const float* __restrict__ ln_g, const float* __restrict__ ln_b,
    float* __restrict__ out) {
  __shared__ float crow[HH];
  __shared__ float red[HH];
  __shared__ float lowv[BD];
  int t = threadIdx.x;
  int b = blockIdx.x >> 10;
  int n = blockIdx.x & (NN - 1);
  size_t row = ((size_t)b * NN + n) * HH;

  crow[t] = ctx[row + t];
  __syncthreads();
  if (t < BD) {
    float s = b_ol[t];
    const float* w = W_ol + (size_t)t * HH;
    for (int j = 0; j < HH; ++j) s += crow[j] * w[j];
    lowv[t] = s;
  }
  __syncthreads();
  float o = b_oh[t];
#pragma unroll
  for (int r = 0; r < BD; ++r) o += lowv[r] * W_oh[(size_t)t * BD + r];
  o += x[row + t];

  red[t] = o;
  __syncthreads();
  for (int s = 128; s > 0; s >>= 1) { if (t < s) red[t] += red[t + s]; __syncthreads(); }
  float mu = red[0] * (1.0f / HH);
  __syncthreads();
  float d = o - mu;
  red[t] = d * d;
  __syncthreads();
  for (int s = 128; s > 0; s >>= 1) { if (t < s) red[t] += red[t + s]; __syncthreads(); }
  float var = red[0] * (1.0f / HH);
  out[row + t] = d * rsqrtf(var + LN_EPS) * ln_g[t] + ln_b[t];

  // reg = REG_W * mean(|attn|); softmax rows are >=0 and sum to 1 => mean = 1/N
  if (blockIdx.x == 0 && t == 0) out[(size_t)BB * NN * HH] = REG_W / (float)NN;
}

// ---------------------------------------------------------------------------
extern "C" void kernel_launch(void* const* d_in, const int* in_sizes, int n_in,
                              void* d_out, int out_size, void* d_ws, size_t ws_size,
                              hipStream_t stream) {
  const float* x       = (const float*)d_in[0];
  const float* loc_emb = (const float*)d_in[1];
  const float* ds_emb  = (const float*)d_in[2];
  const float* W_loc   = (const float*)d_in[3];
  const float* b_loc   = (const float*)d_in[4];
  const float* W_ql    = (const float*)d_in[5];
  const float* b_ql    = (const float*)d_in[6];
  const float* W_qh    = (const float*)d_in[7];
  const float* b_qh    = (const float*)d_in[8];
  const float* u       = (const float*)d_in[9];
  const float* v       = (const float*)d_in[10];
  const float* W_ol    = (const float*)d_in[11];
  const float* b_ol    = (const float*)d_in[12];
  const float* W_oh    = (const float*)d_in[13];
  const float* b_oh    = (const float*)d_in[14];
  const float* ln_g    = (const float*)d_in[15];
  const float* ln_b    = (const float*)d_in[16];
  float* out = (float*)d_out;

  char* ws = (char*)d_ws;
  float*    adj = (float*)(ws);                                // 16 MB
  _Float16* qb  = (_Float16*)(ws + ((size_t)16 << 20));        //  8 MB
  _Float16* kb  = (_Float16*)(ws + ((size_t)24 << 20));        //  8 MB
  _Float16* vtb = (_Float16*)(ws + ((size_t)32 << 20));        //  8 MB
  float*    ctx = (float*)(ws + ((size_t)40 << 20));           // 16 MB

  adj_kernel<<<dim3(HEADS * NN), dim3(256), 0, stream>>>(u, v, adj);
  qkv_kernel<<<dim3(BB * NN / 4), dim3(256), 0, stream>>>(
      x, loc_emb, ds_emb, W_loc, b_loc, W_ql, b_ql, W_qh, b_qh, qb, kb, vtb);
  attn_kernel<<<dim3(BB * HEADS * (NN / 64)), dim3(128), 0, stream>>>(
      qb, kb, vtb, adj, ctx);
  outln_kernel<<<dim3(BB * NN), dim3(256), 0, stream>>>(
      ctx, x, W_ol, b_ol, W_oh, b_oh, ln_g, ln_b, out);
}
